// Net_27462020890736
// MI455X (gfx1250) — matrix operand order repaired and checked by reading, not compiled, not run on hardware
//
#include <hip/hip_runtime.h>

typedef __attribute__((ext_vector_type(16))) _Float16 v16h;
typedef __attribute__((ext_vector_type(8)))  float    v8f;
typedef __attribute__((ext_vector_type(4)))  unsigned int u32x4_t;
typedef __attribute__((ext_vector_type(4)))  int i32x4_t;
typedef __attribute__((ext_vector_type(8)))  int i32x8_t;
typedef _Float16 h16;

constexpr int DN   = 12;     // feature dim
constexpr int PS   = 4;      // flow steps
constexpr int BS   = 16384;  // batch
constexpr int TIL  = 16;     // samples per wave (WMMA M)
constexpr int WV1  = 4;      // waves per block, kernel 1 (fills all 4 SIMD32s of a WGP)
constexpr int WV2  = 8;      // waves per block, kernel 2

__device__ __forceinline__ float sigm(float x) {
  // v_exp_f32 + v_rcp_f32 instead of IEEE divide expansion
  return __builtin_amdgcn_rcpf(1.f + __expf(-x));
}

// C[16][N] = A[16][K] * B[K][N].
// A: f16 LDS row-major (lda halves). Bt: f16 LDS = B^T row-major [N][K] (ldb halves).
// C: f32 LDS (ldc floats). K multiple of 32, N multiple of 16. Whole wave participates.
// A frag (16-bit A 16x32): lanes 0-15 row M=lane hold K {0..15}; lanes 16-31 K {16..31}.
// B frag (16-bit B 32x16): lane L holds col N=L&15, K {0..15} (L<16) or {16..31} (L>=16).
__device__ __forceinline__ void gemm16(const h16* __restrict__ A, int lda,
                                       const h16* __restrict__ Bt, int ldb,
                                       float* __restrict__ C, int ldc,
                                       int N, int K, int lane)
{
  const int r    = lane & 15;
  const int offa = (lane < 16) ? 0 : 16;
  const int offb = (lane < 16) ? 0 : 16;
  for (int nt = 0; nt < N; nt += 16) {
    v8f acc = {};
    for (int k0 = 0; k0 < K; k0 += 32) {
      v16h a, b;
      const h16* ap = A + r * lda + k0 + offa;
      const h16* bp = Bt + (nt + r) * ldb + k0 + offb;
#pragma unroll
      for (int i = 0; i < 16; ++i) { a[i] = ap[i]; }
#pragma unroll
      for (int i = 0; i < 16; ++i) { b[i] = bp[i]; }
      acc = __builtin_amdgcn_wmma_f32_16x16x32_f16(false, a, false, b,
                                                   (short)0, acc, false, false);
    }
    const int mb = (lane < 16) ? 0 : 8;
#pragma unroll
    for (int q = 0; q < 8; ++q) C[(mb + q) * ldc + nt + r] = acc[q];
  }
}

// TDM: async-copy nelem f32 (contiguous) from global to LDS via a 1-D tensor descriptor.
// D# per CDNA5 ISA ch.8: group0 = {count, lds_addr, global_addr, type=2}, group1 = dims.
__device__ __forceinline__ void tdm_load_f32(const float* gsrc, void* ldst, int nelem)
{
  unsigned lds_off = (unsigned)(unsigned long long)ldst;  // LDS generic ptr low 32 = offset
  unsigned long long ga = (unsigned long long)gsrc;
  u32x4_t g0;
  g0[0] = 1u;                                              // count=1 (valid), user mode
  g0[1] = lds_off;                                         // lds_addr [63:32]
  g0[2] = (unsigned)(ga & 0xffffffffu);                    // global_addr [95:64]
  g0[3] = (unsigned)((ga >> 32) & 0x01ffffffu)             // global_addr [120:96]
        | 0x80000000u;                                     // type=2 ("image") [127:126]
  i32x8_t g1;
  g1[0] = (2 << 16);                                       // data_size=2 -> 4 bytes
  g1[1] = (int)(((unsigned)nelem & 0xffffu) << 16);        // tensor_dim0 lo16 [63:48]
  g1[2] = (int)((((unsigned)nelem >> 16) & 0xffffu)        // tensor_dim0 hi16 [79:64]
        | (1u << 16));                                     // tensor_dim1=1    [95:80]
  g1[3] = (int)(((unsigned)nelem & 0xffffu) << 16);        // tile_dim0 [127:112]
  g1[4] = 1;                                               // tile_dim1=1 [143:128]
  g1[5] = nelem;                                           // tensor_dim0_stride lo32
  g1[6] = 0;
  g1[7] = 0;
  i32x4_t gz = {0, 0, 0, 0};
#if defined(__clang_major__) && (__clang_major__ >= 23)
  i32x8_t gz8 = {0, 0, 0, 0, 0, 0, 0, 0};
  __builtin_amdgcn_tensor_load_to_lds(g0, g1, gz, gz, gz8, 0);
#else
  __builtin_amdgcn_tensor_load_to_lds(g0, g1, gz, gz, 0);
#endif
}

struct Ctx {
  // shared weight staging (LDS, f16; "t" = B^T of forward weight, "r" = row-major original)
  const h16 *W1t, *W1r, *W2t, *W2r, *W3t, *W3r;
  const float *b1, *b2, *b3;
  // per-wave scratch (LDS)
  float *zs;   // [PS+1][16][12]
  h16   *s1;   // [16][64]   sigmoid(layer1), f16
  h16   *s2;   // [16][128]  sigmoid(layer2), f16
  h16   *Hb;   // [16][144]  H output, f16
  float *m2b;  // [16][128]  tangent mid
  float *Cw;   // [16][144]  generic GEMM output
  float *d1b;  // [16][64]
  float *d2b;  // [16][128]
  float *gv;   // [16][16]   running grad wrt z
  float *vd;   // [16][16]   temp
  h16   *Af;   // [16][160]  generic f16 A staging
  int lane;
};

// forward h_net at z (16x12 f32 in LDS) -> s1, s2, Hb
__device__ void hnet_forward(Ctx& c, const float* z)
{
  const int lane = c.lane;
  for (int idx = lane; idx < 16 * 32; idx += 32) {
    int rr = idx >> 5, cc = idx & 31;
    c.Af[idx] = (cc < DN) ? (h16)z[rr * DN + cc] : (h16)0.f;
  }
  gemm16(c.Af, 32, c.W1t, 32, c.Cw, 64, 64, 32, lane);
  for (int idx = lane; idx < 16 * 64; idx += 32) {
    float s = sigm(c.Cw[idx] + c.b1[idx & 63]);
    c.s1[idx] = (h16)s;
    c.Af[idx] = (h16)s;
  }
  gemm16(c.Af, 64, c.W2t, 64, c.Cw, 128, 128, 64, lane);
  for (int idx = lane; idx < 16 * 128; idx += 32) {
    float s = sigm(c.Cw[idx] + c.b2[idx & 127]);
    c.s2[idx] = (h16)s;
    c.Af[idx] = (h16)s;
  }
  gemm16(c.Af, 128, c.W3t, 128, c.Cw, 144, 144, 128, lane);
  for (int idx = lane; idx < 16 * 144; idx += 32)
    c.Hb[idx] = (h16)(c.Cw[idx] + c.b3[idx % 144]);
}

// tangent JVP dir k at current point: m2b = (d1*W1[k,:])@W2, Cw = T_k = (d2*m2)@W3
__device__ void tangent_fwd(Ctx& c, int k)
{
  const int lane = c.lane;
  for (int idx = lane; idx < 16 * 64; idx += 32) {
    float s = (float)c.s1[idx];
    c.Af[idx] = (h16)(s * (1.f - s) * (float)c.W1r[k * 64 + (idx & 63)]);
  }
  gemm16(c.Af, 64, c.W2t, 64, c.m2b, 128, 128, 64, lane);
  for (int idx = lane; idx < 16 * 128; idx += 32) {
    float s = (float)c.s2[idx];
    c.Af[idx] = (h16)(s * (1.f - s) * c.m2b[idx]);
  }
  gemm16(c.Af, 128, c.W3t, 128, c.Cw, 144, 144, 128, lane);
}

// per-sample 12x12 Gauss-Jordan with partial pivoting, in place on the ws slot:
// reads J from JG, overwrites it with G = -(J^-1)^T
__device__ void invert_inplace(Ctx& c, float* JG, long gstride)
{
  if (c.lane < 16) {
    float a[DN * DN], inv[DN * DN];
    float* Gp = JG + (long)c.lane * gstride;
    for (int i = 0; i < DN * DN; ++i) { a[i] = Gp[i]; inv[i] = 0.f; }
    for (int i = 0; i < DN; ++i) inv[i * DN + i] = 1.f;
    for (int col = 0; col < DN; ++col) {
      int piv = col; float pv = fabsf(a[col * DN + col]);
      for (int rr = col + 1; rr < DN; ++rr) {
        float v = fabsf(a[rr * DN + col]);
        if (v > pv) { pv = v; piv = rr; }
      }
      if (piv != col)
        for (int cc = 0; cc < DN; ++cc) {
          float t = a[col * DN + cc]; a[col * DN + cc] = a[piv * DN + cc]; a[piv * DN + cc] = t;
          t = inv[col * DN + cc]; inv[col * DN + cc] = inv[piv * DN + cc]; inv[piv * DN + cc] = t;
        }
      float s = __builtin_amdgcn_rcpf(a[col * DN + col]);
      for (int cc = 0; cc < DN; ++cc) { a[col * DN + cc] *= s; inv[col * DN + cc] *= s; }
      for (int rr = 0; rr < DN; ++rr)
        if (rr != col) {
          float f = a[rr * DN + col];
          for (int cc = 0; cc < DN; ++cc) {
            a[rr * DN + cc]   -= f * a[col * DN + cc];
            inv[rr * DN + cc] -= f * inv[col * DN + cc];
          }
        }
    }
    for (int j = 0; j < DN; ++j)
      for (int k = 0; k < DN; ++k)
        Gp[j * DN + k] = -inv[k * DN + j];
  }
}

// accumulate into gv the gradient of -log|det J| wrt evaluation point v (uses stored G)
__device__ void jacgrad(Ctx& c, const float* v, const float* Gt, long GS)
{
  const int lane = c.lane;
  hnet_forward(c, v);
  for (int idx = lane; idx < 16 * 64; idx += 32) c.d1b[idx] = 0.f;
  for (int idx = lane; idx < 16 * 128; idx += 32) c.d2b[idx] = 0.f;
  for (int idx = lane; idx < 16 * 16; idx += 32) c.vd[idx] = 0.f;
  for (int k = 0; k < DN; ++k) {
    tangent_fwd(c, k);                       // m2b = m2_k, Cw = T_k
    if (lane < 16) {                         // explicit v-term: vd_i += sum_j G[j,k] T[12i+j]
      const float* G = Gt + (long)lane * GS;
      const float* T = c.Cw + lane * 144;
      for (int i = 0; i < DN; ++i) {
        float acc = 0.f;
        for (int j = 0; j < DN; ++j) acc += G[j * DN + k] * T[DN * i + j];
        c.vd[lane * 16 + i] += acc;
      }
    }
    // Tbar_k[12i+j] = G[j,k]*v_i  -> u2bar = Tbar @ W3^T
    for (int idx = lane; idx < 16 * 160; idx += 32) {
      int rr = idx / 160, cc = idx - rr * 160;
      float val = 0.f;
      if (cc < 144) {
        int i = cc / DN, j = cc - i * DN;
        val = Gt[(long)rr * GS + j * DN + k] * v[rr * DN + i];
      }
      c.Af[idx] = (h16)val;
    }
    gemm16(c.Af, 160, c.W3r, 160, c.Cw, 128, 128, 160, lane);
    for (int idx = lane; idx < 16 * 128; idx += 32) {
      float ub = c.Cw[idx];
      c.d2b[idx] += ub * c.m2b[idx];
      float s = (float)c.s2[idx];
      c.Af[idx] = (h16)(ub * s * (1.f - s));            // m2bar
    }
    gemm16(c.Af, 128, c.W2r, 128, c.Cw, 64, 64, 128, lane);  // u1bar
    for (int idx = lane; idx < 16 * 64; idx += 32)
      c.d1b[idx] += c.Cw[idx] * (float)c.W1r[k * 64 + (idx & 63)];
  }
  // Hbar = G^T path + sigmoid'' terms
  for (int idx = lane; idx < 16 * 160; idx += 32) {
    int rr = idx / 160, cc = idx - rr * 160;
    float val = 0.f;
    if (cc < 144) {
      int kk = cc / DN, j = cc - kk * DN;
      val = Gt[(long)rr * GS + j * DN + kk];
    }
    c.Af[idx] = (h16)val;
  }
  gemm16(c.Af, 160, c.W3r, 160, c.Cw, 128, 128, 160, lane);  // s2bar (H part)
  for (int idx = lane; idx < 16 * 128; idx += 32) {
    float s = (float)c.s2[idx];
    float s2bar = c.Cw[idx] + c.d2b[idx] * (1.f - 2.f * s);
    c.Af[idx] = (h16)(s2bar * s * (1.f - s));               // a2bar
  }
  gemm16(c.Af, 128, c.W2r, 128, c.Cw, 64, 64, 128, lane);   // s1bar (net part)
  for (int idx = lane; idx < 16 * 64; idx += 32) {
    float s = (float)c.s1[idx];
    float s1bar = c.Cw[idx] + c.d1b[idx] * (1.f - 2.f * s);
    c.Af[idx] = (h16)(s1bar * s * (1.f - s));               // a1bar
  }
  gemm16(c.Af, 64, c.W1r, 64, c.Cw, 16, 16, 64, lane);      // vbar_net [16][16]
  if (lane < 16)
    for (int i = 0; i < DN; ++i)
      c.gv[lane * 16 + i] += c.vd[lane * 16 + i] + c.Cw[lane * 16 + i];
}

// chain gv through z_{t+1} = z_t @ H(z_t):  g <- g@H^T + hnet-backprop(z_t (x) g)
__device__ void chain_step(Ctx& c, const float* zt)
{
  const int lane = c.lane;
  hnet_forward(c, zt);
  if (lane < 16) {
    const float* g = c.gv + lane * 16;
    const h16*   H = c.Hb + lane * 144;
    for (int i = 0; i < DN; ++i) {
      float acc = 0.f;
      for (int j = 0; j < DN; ++j) acc += g[j] * (float)H[i * DN + j];
      c.vd[lane * 16 + i] = acc;
    }
  }
  for (int idx = lane; idx < 16 * 160; idx += 32) {  // Hbar[i*12+j] = z_i * g_j
    int rr = idx / 160, cc = idx - rr * 160;
    float val = 0.f;
    if (cc < 144) {
      int i = cc / DN, j = cc - i * DN;
      val = zt[rr * DN + i] * c.gv[rr * 16 + j];
    }
    c.Af[idx] = (h16)val;
  }
  gemm16(c.Af, 160, c.W3r, 160, c.Cw, 128, 128, 160, lane);
  for (int idx = lane; idx < 16 * 128; idx += 32) {
    float s = (float)c.s2[idx];
    c.Af[idx] = (h16)(c.Cw[idx] * s * (1.f - s));
  }
  gemm16(c.Af, 128, c.W2r, 128, c.Cw, 64, 64, 128, lane);
  for (int idx = lane; idx < 16 * 64; idx += 32) {
    float s = (float)c.s1[idx];
    c.Af[idx] = (h16)(c.Cw[idx] * s * (1.f - s));
  }
  gemm16(c.Af, 64, c.W1r, 64, c.Cw, 16, 16, 64, lane);
  if (lane < 16)
    for (int i = 0; i < DN; ++i)
      c.gv[lane * 16 + i] = c.vd[lane * 16 + i] + c.Cw[lane * 16 + i];
}

__global__ void __launch_bounds__(WV1 * 32) flow_kernel(
    const float* __restrict__ x,
    const float* __restrict__ W1, const float* __restrict__ b1,
    const float* __restrict__ W2, const float* __restrict__ b2,
    const float* __restrict__ W3, const float* __restrict__ b3,
    float* __restrict__ out, float* __restrict__ Gws, float* __restrict__ zws)
{
  extern __shared__ char smem[];
  char* p = smem;
  auto carve = [&](size_t b) { char* q = p; p += (b + 15) & ~size_t(15); return q; };
  h16* W1t = (h16*)carve(64 * 32 * 2);
  h16* W1r = (h16*)carve(16 * 64 * 2);
  h16* W2t = (h16*)carve(128 * 64 * 2);
  h16* W2r = (h16*)carve(64 * 128 * 2);
  h16* W3t = (h16*)carve(144 * 128 * 2);
  h16* W3r = (h16*)carve(128 * 160 * 2);
  float* sb1 = (float*)carve(64 * 4);
  float* sb2 = (float*)carve(128 * 4);
  float* sb3 = (float*)carve(144 * 4);
  float* zsA = (float*)carve(WV1 * (PS + 1) * TIL * DN * 4);
  h16*   s1A = (h16*)carve(WV1 * 16 * 64 * 2);
  h16*   s2A = (h16*)carve(WV1 * 16 * 128 * 2);
  h16*   HbA = (h16*)carve(WV1 * 16 * 144 * 2);
  float* m2A = (float*)carve(WV1 * 16 * 128 * 4);
  float* CwA = (float*)carve(WV1 * 16 * 144 * 4);
  float* d1A = (float*)carve(WV1 * 16 * 64 * 4);
  float* d2A = (float*)carve(WV1 * 16 * 128 * 4);
  float* gvA = (float*)carve(WV1 * 16 * 16 * 4);
  float* vdA = (float*)carve(WV1 * 16 * 16 * 4);
  h16*   AfA = (h16*)carve(WV1 * 16 * 160 * 2);

  const int tid = threadIdx.x, nthr = blockDim.x;
  __builtin_prefetch(W3, 0, 0);
  __builtin_prefetch(W2, 0, 0);
  for (int i = tid; i < 64 * 32; i += nthr) { int n = i >> 5, k = i & 31; W1t[i] = (k < DN) ? (h16)W1[k * 64 + n] : (h16)0.f; }
  for (int i = tid; i < 16 * 64; i += nthr) { int r = i >> 6, cc = i & 63; W1r[i] = (r < DN) ? (h16)W1[r * 64 + cc] : (h16)0.f; }
  for (int i = tid; i < 128 * 64; i += nthr) { int n = i >> 6, k = i & 63; W2t[i] = (h16)W2[k * 128 + n]; }
  for (int i = tid; i < 64 * 128; i += nthr) W2r[i] = (h16)W2[i];
  for (int i = tid; i < 144 * 128; i += nthr) { int n = i >> 7, k = i & 127; W3t[i] = (h16)W3[k * 144 + n]; }
  for (int i = tid; i < 128 * 160; i += nthr) { int rI = i / 160, cI = i - rI * 160; W3r[i] = (cI < 144) ? (h16)W3[rI * 144 + cI] : (h16)0.f; }
  for (int i = tid; i < 64; i += nthr) sb1[i] = b1[i];
  for (int i = tid; i < 128; i += nthr) sb2[i] = b2[i];
  for (int i = tid; i < 144; i += nthr) sb3[i] = b3[i];

  const int wave = tid >> 5, lane = tid & 31;
  Ctx c;
  c.lane = lane;
  c.W1t = W1t; c.W1r = W1r; c.W2t = W2t; c.W2r = W2r; c.W3t = W3t; c.W3r = W3r;
  c.b1 = sb1; c.b2 = sb2; c.b3 = sb3;
  c.zs  = zsA + wave * (PS + 1) * TIL * DN;
  c.s1  = s1A + wave * 16 * 64;
  c.s2  = s2A + wave * 16 * 128;
  c.Hb  = HbA + wave * 16 * 144;
  c.m2b = m2A + wave * 16 * 128;
  c.Cw  = CwA + wave * 16 * 144;
  c.d1b = d1A + wave * 16 * 64;
  c.d2b = d2A + wave * 16 * 128;
  c.gv  = gvA + wave * 16 * 16;
  c.vd  = vdA + wave * 16 * 16;
  c.Af  = AfA + wave * 16 * 160;

  const long tilebase = ((long)blockIdx.x * WV1 + wave) * TIL;
  const long GS = (long)PS * DN * DN;  // per-sample stride in Gws

  // async TDM copy of this wave's x tile (16x12 f32, contiguous) into zs
  tdm_load_f32(x + tilebase * DN, c.zs, TIL * DN);
  __syncthreads();                         // weights visible to all waves
  __builtin_amdgcn_s_wait_tensorcnt(0);    // x tile landed in LDS

  hnet_forward(c, c.zs);  // acts at z0 (invariant: s1,s2,Hb at z_t at loop top)

  for (int t = 0; t < PS; ++t) {
    float* zt = c.zs + t * TIL * DN;
    float* zn = c.zs + (t + 1) * TIL * DN;
    if (lane < 16) {                       // z_{t+1} = z_t @ H_t
      const float* zr = zt + lane * DN;
      const h16*   H  = c.Hb + lane * 144;
      for (int j = 0; j < DN; ++j) {
        float acc = 0.f;
        for (int i = 0; i < DN; ++i) acc += zr[i] * (float)H[i * DN + j];
        zn[lane * DN + j] = acc;
      }
    }
    hnet_forward(c, zn);                   // acts + H at z_{t+1} (Jacobian point)
    float* JGbase = Gws + tilebase * GS + t * DN * DN;   // J assembled straight into ws
    for (int k = 0; k < DN; ++k) {
      tangent_fwd(c, k);                   // Cw = T_k
      if (lane < 16) {                     // J[j,k] = H[k,j] + sum_i v_i T_k[12i+j]
        const float* T  = c.Cw + lane * 144;
        const float* vz = zn + lane * DN;
        const h16*   Hs = c.Hb + lane * 144;
        float* Js = JGbase + (long)lane * GS;
        for (int j = 0; j < DN; ++j) {
          float acc = 0.f;
          for (int i = 0; i < DN; ++i) acc += vz[i] * T[i * DN + j];
          Js[j * DN + k] = (float)Hs[k * DN + j] + acc;
        }
      }
    }
    invert_inplace(c, JGbase, GS);         // J -> G = -(J^-1)^T, in place
  }

  for (int idx = lane; idx < TIL * DN; idx += 32)       // z4 for x_net kernel
    zws[tilebase * DN + idx] = c.zs[PS * TIL * DN + idx];

  for (int idx = lane; idx < TIL * 16; idx += 32) c.gv[idx] = 0.f;
  for (int t = PS - 1; t >= 0; --t) {
    jacgrad(c, c.zs + (t + 1) * TIL * DN, Gws + tilebase * GS + t * DN * DN, GS);
    chain_step(c, c.zs + t * TIL * DN);
  }

  for (int idx = lane; idx < TIL * DN; idx += 32) {
    int s = idx / DN, j = idx - s * DN;
    out[(tilebase + s) * 13 + 1 + j] = c.gv[s * 16 + j];
  }
}

__global__ void __launch_bounds__(WV2 * 32) xnet_kernel(
    const float* __restrict__ z,
    const float* __restrict__ XW1, const float* __restrict__ Xb1,
    const float* __restrict__ XW2, const float* __restrict__ Xb2,
    const float* __restrict__ XW3, const float* __restrict__ Xb3,
    const float* __restrict__ XW4, const float* __restrict__ Xb4,
    float* __restrict__ out)
{
  extern __shared__ char smem[];
  char* p = smem;
  auto carve = [&](size_t b) { char* q = p; p += (b + 15) & ~size_t(15); return q; };
  h16* A1 = (h16*)carve(64 * 32 * 2);
  h16* A2 = (h16*)carve(128 * 64 * 2);
  h16* A3 = (h16*)carve(128 * 128 * 2);
  float* w4 = (float*)carve(128 * 4);
  float* c1 = (float*)carve(64 * 4);
  float* c2 = (float*)carve(128 * 4);
  float* c3 = (float*)carve(128 * 4);
  h16*   AfA = (h16*)carve(WV2 * 16 * 128 * 2);
  float* CwA = (float*)carve(WV2 * 16 * 128 * 4);

  const int tid = threadIdx.x, nthr = blockDim.x;
  for (int i = tid; i < 64 * 32; i += nthr) { int n = i >> 5, k = i & 31; A1[i] = (k < DN) ? (h16)XW1[k * 64 + n] : (h16)0.f; }
  for (int i = tid; i < 128 * 64; i += nthr) { int n = i >> 6, k = i & 63; A2[i] = (h16)XW2[k * 128 + n]; }
  for (int i = tid; i < 128 * 128; i += nthr) { int n = i >> 7, k = i & 127; A3[i] = (h16)XW3[k * 128 + n]; }
  for (int i = tid; i < 128; i += nthr) w4[i] = XW4[i];
  for (int i = tid; i < 64; i += nthr) c1[i] = Xb1[i];
  for (int i = tid; i < 128; i += nthr) c2[i] = Xb2[i];
  for (int i = tid; i < 128; i += nthr) c3[i] = Xb3[i];
  __syncthreads();

  const int wave = tid >> 5, lane = tid & 31;
  h16*   Af = AfA + wave * 16 * 128;
  float* Cw = CwA + wave * 16 * 128;
  const long base = ((long)blockIdx.x * WV2 + wave) * TIL;

  for (int idx = lane; idx < 16 * 32; idx += 32) {
    int r = idx >> 5, cc = idx & 31;
    Af[idx] = (cc < DN) ? (h16)z[(base + r) * DN + cc] : (h16)0.f;
  }
  gemm16(Af, 32, A1, 32, Cw, 64, 64, 32, lane);
  for (int idx = lane; idx < 16 * 64; idx += 32)
    Af[idx] = (h16)fmaxf(Cw[idx] + c1[idx & 63], 0.f);
  gemm16(Af, 64, A2, 64, Cw, 128, 128, 64, lane);
  for (int idx = lane; idx < 16 * 128; idx += 32)
    Af[idx] = (h16)fmaxf(Cw[idx] + c2[idx & 127], 0.f);
  gemm16(Af, 128, A3, 128, Cw, 128, 128, 128, lane);
  for (int idx = lane; idx < 16 * 128; idx += 32)
    Cw[idx] = fmaxf(Cw[idx] + c3[idx & 127], 0.f);
  if (lane < 16) {
    float acc = Xb4[0];
    const float* h = Cw + lane * 128;
    for (int i = 0; i < 128; ++i) acc += h[i] * w4[i];
    out[(base + lane) * 13] = sigm(acc);
  }
}

extern "C" void kernel_launch(void* const* d_in, const int* in_sizes, int n_in,
                              void* d_out, int out_size, void* d_ws, size_t ws_size,
                              hipStream_t stream)
{
  const float* x   = (const float*)d_in[0];
  const float* W1  = (const float*)d_in[1];
  const float* b1  = (const float*)d_in[2];
  const float* W2  = (const float*)d_in[3];
  const float* b2  = (const float*)d_in[4];
  const float* W3  = (const float*)d_in[5];
  const float* b3  = (const float*)d_in[6];
  const float* XW1 = (const float*)d_in[7];
  const float* Xb1 = (const float*)d_in[8];
  const float* XW2 = (const float*)d_in[9];
  const float* Xb2 = (const float*)d_in[10];
  const float* XW3 = (const float*)d_in[11];
  const float* Xb3 = (const float*)d_in[12];
  const float* XW4 = (const float*)d_in[13];
  const float* Xb4 = (const float*)d_in[14];
  float* out = (float*)d_out;
  float* Gws = (float*)d_ws;                       // BS * PS * 144 floats: J then -J^-T per step
  float* zws = Gws + (size_t)BS * PS * DN * DN;    // BS * 12 floats (z4)

  // LDS budgets (all sub-arrays are 16B multiples, so no carve slack)
  constexpr int SH1 =
      (64*32 + 16*64 + 128*64 + 64*128 + 144*128 + 128*160) * 2 + (64 + 128 + 144) * 4 +
      WV1 * ((PS+1)*TIL*DN*4 + 16*64*2 + 16*128*2 + 16*144*2 + 16*128*4 +
             16*144*4 + 16*64*4 + 16*128*4 + 16*16*4 + 16*16*4 + 16*160*2);
  constexpr int SH2 =
      (64*32 + 128*64 + 128*128) * 2 + (128 + 64 + 128 + 128) * 4 +
      WV2 * (16*128*2 + 16*128*4);
  static_assert(SH1 <= 320 * 1024, "kernel1 LDS exceeds WGP capacity");

  (void)hipFuncSetAttribute((const void*)flow_kernel,
                            hipFuncAttributeMaxDynamicSharedMemorySize, SH1);
  (void)hipFuncSetAttribute((const void*)xnet_kernel,
                            hipFuncAttributeMaxDynamicSharedMemorySize, SH2);

  flow_kernel<<<BS / (WV1 * TIL), WV1 * 32, SH1, stream>>>(
      x, W1, b1, W2, b2, W3, b3, out, Gws, zws);
  xnet_kernel<<<BS / (WV2 * TIL), WV2 * 32, SH2, stream>>>(
      zws, XW1, Xb1, XW2, Xb2, XW3, Xb3, XW4, Xb4, out);
}
